// CNN_3D_6828998001195
// MI455X (gfx1250) — compile-verified, matrix operand order checked by reading, not run on hardware
//
#include <hip/hip_runtime.h>
#include <math.h>

// ---------------- problem constants ----------------
#define BB    32
#define SS    200
#define KK    8
#define HH    225
#define EE    200
#define CLIPS 193
#define NCNN  (BB*CLIPS)             // 6176
#define NCNNP 6208                   // padded to multiple of 64
#define NPOS  ((long long)NCNN*1800) // 11,116,800 spatial positions per channel
#define G4H   900                    // 4*H
#define NPAD  960                    // padded rows of B matrices (mult of 64)
#define ASTR  40                     // LDS row stride (ushorts) to dodge bank conflicts

#if defined(__gfx1250__)
typedef __attribute__((ext_vector_type(16))) __bf16 bfx16;
typedef __attribute__((ext_vector_type(8)))  __bf16 bfx8;
typedef __attribute__((ext_vector_type(8)))  float  fx8;

__device__ __forceinline__ bfx16 ldfrag(const unsigned short* base, int roff, int kb0, int kb1) {
  bfx8 lo = *(const bfx8*)(base + roff + kb0);
  bfx8 hv = *(const bfx8*)(base + roff + kb1);
  return __builtin_shufflevector(lo, hv, 0,1,2,3,4,5,6,7,8,9,10,11,12,13,14,15);
}

// async DMA: global (16B) -> LDS, tracked by ASYNCcnt (no VGPR round trip)
__device__ __forceinline__ void async_b128(unsigned int lds_off, const void* gptr) {
  unsigned long long ga = (unsigned long long)gptr;
  asm volatile("global_load_async_to_lds_b128 %0, %1, off" :: "v"(lds_off), "v"(ga) : "memory");
}
__device__ __forceinline__ void wait_async0() {
  asm volatile("s_wait_asynccnt 0x0" ::: "memory");
}
#endif

// fp32 -> bf16 (round to nearest even), portable for host parse
__device__ __forceinline__ unsigned short f2bf(float f) {
  union { float f; unsigned int u; } v; v.f = f;
  unsigned int u = v.u;
  unsigned int r = (u + 0x7FFFu + ((u >> 16) & 1u)) >> 16;
  return (unsigned short)r;
}

// ---------------- small elementwise kernels ----------------
__global__ void zero_f(float* p, int n) {
  int i = blockIdx.x * blockDim.x + threadIdx.x;
  if (i < n) p[i] = 0.f;
}

__global__ void bias_sum(const float* a, const float* b, float* o, int n) {
  int i = blockIdx.x * blockDim.x + threadIdx.x;
  if (i < n) o[i] = a[i] + b[i];
}

// embedding gather -> bf16 padded [6400][224]
__global__ void embed_bf(const int* __restrict__ x, const float* __restrict__ emb,
                         unsigned short* __restrict__ xe) {
  int idx = blockIdx.x * blockDim.x + threadIdx.x;
  if (idx >= BB * SS * 224) return;
  int r = idx / 224, k = idx % 224;
  xe[idx] = (k < EE) ? f2bf(emb[(long long)x[r] * EE + k]) : (unsigned short)0;
}

// generic fp32[M][Ksz] -> bf16 [Mp][Kp] zero-padded
__global__ void cvt_pad(const float* __restrict__ src, unsigned short* __restrict__ dst,
                        int M, int Ksz, int Kp, long long total) {
  long long idx = (long long)blockIdx.x * blockDim.x + threadIdx.x;
  if (idx >= total) return;
  int r = (int)(idx / Kp), c = (int)(idx % Kp);
  dst[idx] = (r < M && c < Ksz) ? f2bf(src[(long long)r * Ksz + c]) : (unsigned short)0;
}

// conv weights [Co][Ci*27] -> bf16 [16][448] zero-padded
__global__ void pack_w(const float* __restrict__ w, unsigned short* __restrict__ dst,
                       int Co, int Ktot) {
  int idx = blockIdx.x * blockDim.x + threadIdx.x;
  if (idx >= 16 * 448) return;
  int co = idx / 448, k = idx % 448;
  dst[idx] = (co < Co && k < Ktot) ? f2bf(w[co * Ktot + k]) : (unsigned short)0;
}

// ---------------- bf16 WMMA GEMM: C[M][N] = A[M,Kp] * B[N,Kp]^T (+bias) ----------------
// 128 threads = 4 waves, each wave owns a 32x32 C tile (2x2 WMMA frags) -> 64x64 per block.
// Double-buffered LDS staging via global_load_async_to_lds_b128 (ASYNCcnt), 1 barrier/chunk.
// A buffer rows mult of 64, B buffer rows = NPAD, Kp mult of 32.
__global__ void gemm_bf16_nt(const unsigned short* __restrict__ A,
                             const unsigned short* __restrict__ Bm,
                             const float* __restrict__ bias, float* __restrict__ C,
                             int M, int N, int Kp, int ldc) {
#if defined(__gfx1250__)
  __shared__ __align__(16) unsigned short As[2][64 * ASTR];
  __shared__ __align__(16) unsigned short Bs[2][64 * ASTR];
  int m0 = blockIdx.x * 64, n0 = blockIdx.y * 64;
  int t = threadIdx.x;
  int wave = t >> 5, lane = t & 31;
  int wm = wave >> 1, wn = wave & 1;
  int hi = (lane >= 16);
  int l15 = lane & 15;
  int nk = Kp >> 5;

  // stage chunk 0 into buffer 0 (async DMA straight to LDS)
#pragma unroll
  for (int q = 0; q < 2; ++q) {
    int c2 = q * 128 + t;
    int row = c2 >> 2, seg = c2 & 3;
    async_b128((unsigned int)(size_t)&As[0][row * ASTR + seg * 8],
               &A[(long long)(m0 + row) * Kp + seg * 8]);
    async_b128((unsigned int)(size_t)&Bs[0][row * ASTR + seg * 8],
               &Bm[(long long)(n0 + row) * Kp + seg * 8]);
  }
  wait_async0();
  __syncthreads();

  fx8 acc00 = {}, acc01 = {}, acc10 = {}, acc11 = {};
  for (int kc = 0; kc < nk; ++kc) {
    int cur = kc & 1;
    if (kc + 1 < nk) {   // prefetch next chunk while computing this one
      int k0 = (kc + 1) * 32;
#pragma unroll
      for (int q = 0; q < 2; ++q) {
        int c2 = q * 128 + t;
        int row = c2 >> 2, seg = c2 & 3;
        async_b128((unsigned int)(size_t)&As[cur ^ 1][row * ASTR + seg * 8],
                   &A[(long long)(m0 + row) * Kp + k0 + seg * 8]);
        async_b128((unsigned int)(size_t)&Bs[cur ^ 1][row * ASTR + seg * 8],
                   &Bm[(long long)(n0 + row) * Kp + k0 + seg * 8]);
      }
    }
    int kb0 = hi ? 8 : 0, kb1 = hi ? 24 : 16;
    bfx16 a0 = ldfrag(&As[cur][0], (wm * 32 + l15) * ASTR,      kb0, kb1);
    bfx16 a1 = ldfrag(&As[cur][0], (wm * 32 + 16 + l15) * ASTR, kb0, kb1);
    bfx16 b0 = ldfrag(&Bs[cur][0], (wn * 32 + l15) * ASTR,      kb0, kb1);
    bfx16 b1 = ldfrag(&Bs[cur][0], (wn * 32 + 16 + l15) * ASTR, kb0, kb1);
    acc00 = __builtin_amdgcn_wmma_f32_16x16x32_bf16(false, a0, false, b0, (short)0, acc00, false, false);
    acc01 = __builtin_amdgcn_wmma_f32_16x16x32_bf16(false, a0, false, b1, (short)0, acc01, false, false);
    acc10 = __builtin_amdgcn_wmma_f32_16x16x32_bf16(false, a1, false, b0, (short)0, acc10, false, false);
    acc11 = __builtin_amdgcn_wmma_f32_16x16x32_bf16(false, a1, false, b1, (short)0, acc11, false, false);
    wait_async0();
    __syncthreads();
  }

  // epilogue: C layout 16x16 f32 per frag (lanes 0-15: N=lane,M=r; 16-31: M=8+r)
#pragma unroll
  for (int i = 0; i < 2; ++i) {
#pragma unroll
    for (int j = 0; j < 2; ++j) {
      fx8 acc = (i == 0) ? (j == 0 ? acc00 : acc01) : (j == 0 ? acc10 : acc11);
      int n = n0 + wn * 32 + j * 16 + l15;
      int mb = m0 + wm * 32 + i * 16 + (hi ? 8 : 0);
      if (n < N) {
        float bv = bias ? bias[n] : 0.f;
#pragma unroll
        for (int r2 = 0; r2 < 8; ++r2) {
          int m = mb + r2;
          if (m < M) C[(long long)m * ldc + n] = acc[r2] + bv;
        }
      }
    }
  }
#endif
}

// ---------------- LSTM recurrence pieces ----------------
// hbf is padded to 64 rows x 256 cols so the 64-wide GEMM tile never reads OOB
__global__ void lstm_init(const float* __restrict__ h0, const float* __restrict__ c0,
                          float* __restrict__ cst, unsigned short* __restrict__ hbf) {
  int idx = blockIdx.x * blockDim.x + threadIdx.x;
  if (idx >= 64 * 256) return;
  int b = idx / 256, j = idx % 256;
  if (b < BB && j < HH) {
    int q = b * HH + j;
    cst[q] = c0[q];
    hbf[idx] = f2bf(h0[q]);
  } else {
    hbf[idx] = 0;
  }
}

// gates i,f,g,o; zpre rows indexed (b*T + t)
__global__ void lstm_step(const float* __restrict__ zpre, const float* __restrict__ zh,
                          float* __restrict__ cst, unsigned short* __restrict__ hbf,
                          float* __restrict__ hout, int t, int T) {
  int idx = blockIdx.x * blockDim.x + threadIdx.x;
  if (idx >= BB * HH) return;
  int b = idx / HH, j = idx % HH;
  const float* zp = zpre + (long long)(b * T + t) * G4H;
  const float* zb = zh + b * G4H;
  float zi = zp[j]          + zb[j];
  float zf = zp[HH + j]     + zb[HH + j];
  float zg = zp[2 * HH + j] + zb[2 * HH + j];
  float zo = zp[3 * HH + j] + zb[3 * HH + j];
  float ig = 1.f / (1.f + expf(-zi));
  float fg = 1.f / (1.f + expf(-zf));
  float gg = tanhf(zg);
  float og = 1.f / (1.f + expf(-zo));
  float cn = fg * cst[idx] + ig * gg;
  cst[idx] = cn;
  float hn = og * tanhf(cn);
  hbf[b * 256 + j] = f2bf(hn);
  hout[(long long)(b * T + t) * HH + j] = hn;
}

// ---------------- implicit-GEMM 3D conv with WMMA ----------------
// A: im2col rows (16 positions / wave), gathered on the fly with fused BN+ReLU of the
// previous layer's raw output. B: packed bf16 weights [16][448] async-staged in LDS.
// zout is channel-major: zout[co*NPOS + p], p = n*1800 + d*225 + h*15 + w.
__global__ void conv3d_wmma(const float* __restrict__ zin, const float* __restrict__ hf,
                            const float* __restrict__ norm, const unsigned short* __restrict__ wpk,
                            float* __restrict__ zout, int Ci, int Co, int Kp, int is_first) {
#if defined(__gfx1250__)
  __shared__ __align__(16) unsigned short wlds[16 * 448];
  for (int i = threadIdx.x; i < 896; i += blockDim.x)   // 896 x 16B = 14336 B
    async_b128((unsigned int)(size_t)&wlds[i * 8], &wpk[i * 8]);
  wait_async0();
  __syncthreads();

  int Ktot = Ci * 27;
  int wave = threadIdx.x >> 5, lane = threadIdx.x & 31;
  int hi = (lane >= 16);
  long long p0 = (long long)blockIdx.x * 64 + wave * 16;

  // decode this lane's im2col row (A-matrix row = lane&15)
  long long p = p0 + (lane & 15);
  int n = (int)(p / 1800);
  int rem = (int)(p % 1800);
  int d = rem / 225; rem %= 225;
  int hhp = rem / 15, wwp = rem % 15;
  int bq = n / CLIPS, clip = n % CLIPS;   // layer-1 gather from hf

  fx8 acc = {};
  for (int kc = 0; kc < Kp; kc += 32) {
    bfx16 a;
#pragma unroll
    for (int j = 0; j < 16; ++j) {
      int koff = (j < 8) ? ((hi ? 8 : 0) + j) : ((hi ? 24 : 16) + (j - 8));
      int k = kc + koff;
      float v = 0.f;
      if (k < Ktot) {
        int ci = k / 27, r = k % 27;
        int kd = r / 9; r %= 9;
        int kh = r / 3, kw = r % 3;
        int id = d + kd - 1, ih = hhp + kh - 1, iw = wwp + kw - 1;
        if ((unsigned)id < 8u && (unsigned)ih < 15u && (unsigned)iw < 15u) {
          if (is_first) {
            v = hf[((long long)(bq * SS + clip + id)) * HH + ih * 15 + iw];
          } else {
            float raw = zin[(long long)ci * NPOS + (long long)n * 1800 + id * 225 + ih * 15 + iw];
            v = fmaxf(raw * norm[ci * 2] + norm[ci * 2 + 1], 0.f);
          }
        }
      }
      a[j] = (__bf16)v;
    }
    int co = lane & 15;
    int kb0 = kc + (hi ? 8 : 0), kb1 = kc + (hi ? 24 : 16);
    bfx8 b0 = *(const bfx8*)&wlds[co * 448 + kb0];
    bfx8 b1 = *(const bfx8*)&wlds[co * 448 + kb1];
    bfx16 bf = __builtin_shufflevector(b0, b1, 0,1,2,3,4,5,6,7,8,9,10,11,12,13,14,15);
    acc = __builtin_amdgcn_wmma_f32_16x16x32_bf16(false, a, false, bf, (short)0, acc,
                                                  false, false);
  }
  int co = lane & 15;
  if (co < Co) {
    long long mbase = p0 + (hi ? 8 : 0);
#pragma unroll
    for (int r2 = 0; r2 < 8; ++r2)
      zout[(long long)co * NPOS + mbase + r2] = acc[r2];
  }
#endif
}

// per-channel sum / sumsq over NPOS elements (channel-major), atomics into stats[c*2(+1)]
__global__ void bn_stats(const float* __restrict__ z, float* __restrict__ stats) {
  int c = blockIdx.y;
  float s = 0.f, ss = 0.f;
  long long stride = (long long)gridDim.x * blockDim.x;
  for (long long i = (long long)blockIdx.x * blockDim.x + threadIdx.x; i < NPOS; i += stride) {
    float v = z[(long long)c * NPOS + i];
    s += v; ss += v * v;
  }
  __shared__ float sh[256], sh2[256];
  sh[threadIdx.x] = s; sh2[threadIdx.x] = ss;
  __syncthreads();
  for (int o = 128; o > 0; o >>= 1) {
    if ((int)threadIdx.x < o) { sh[threadIdx.x] += sh[threadIdx.x + o]; sh2[threadIdx.x] += sh2[threadIdx.x + o]; }
    __syncthreads();
  }
  if (threadIdx.x == 0) {
    atomicAdd(&stats[c * 2], sh[0]);
    atomicAdd(&stats[c * 2 + 1], sh2[0]);
  }
}

__global__ void bn_finalize(const float* __restrict__ stats, const float* __restrict__ g,
                            const float* __restrict__ be, float* __restrict__ norm, int Co) {
  int c = threadIdx.x;
  if (c >= Co) return;
  const float NPf = (float)NPOS;
  float mean = stats[c * 2] / NPf;
  float var = stats[c * 2 + 1] / NPf - mean * mean;
  float rstd = rsqrtf(var + 1e-5f);
  float sc = g[c] * rstd;
  norm[c * 2] = sc;
  norm[c * 2 + 1] = be[c] - mean * sc;
}

// layer-6 BN + tanh + reshape -> bf16 [6208][1824] (rows = b*CLIPS+clip, zero-padded tail)
__global__ void pack_ycnn(const float* __restrict__ z6, const float* __restrict__ norm6,
                          unsigned short* __restrict__ y) {
  long long idx = (long long)blockIdx.x * blockDim.x + threadIdx.x;
  if (idx >= (long long)NCNNP * 1824) return;
  int n = (int)(idx / 1824), k = (int)(idx % 1824);
  if (n < NCNN && k < 1800) {
    float v = z6[(long long)n * 1800 + k];
    y[idx] = f2bf(tanhf(v * norm6[0] + norm6[1]));
  } else {
    y[idx] = 0;
  }
}

__global__ void fc_out(const float* __restrict__ hr, const float* __restrict__ W,
                       const float* __restrict__ bfc, float* __restrict__ out) {
  int pId = blockIdx.x * blockDim.x + threadIdx.x;
  if (pId >= NCNN) return;
  float s = bfc[0];
  const float* row = hr + (long long)pId * HH;
#pragma unroll 5
  for (int j = 0; j < HH; ++j) s += row[j] * W[j];
  out[pId] = s;
}

// ---------------- host orchestration ----------------
extern "C" void kernel_launch(void* const* d_in, const int* in_sizes, int n_in,
                              void* d_out, int out_size, void* d_ws, size_t ws_size,
                              hipStream_t stream) {
  (void)in_sizes; (void)n_in; (void)out_size; (void)ws_size;
  const int*   x     = (const int*)d_in[0];
  const float* emb   = (const float*)d_in[1];
  const float* Wih_f = (const float*)d_in[2];
  const float* Whh_f = (const float*)d_in[3];
  const float* bih_f = (const float*)d_in[4];
  const float* bhh_f = (const float*)d_in[5];
  const float* h0f   = (const float*)d_in[6];
  const float* c0f   = (const float*)d_in[7];
  const float* Wih_r = (const float*)d_in[8];
  const float* Whh_r = (const float*)d_in[9];
  const float* bih_r = (const float*)d_in[10];
  const float* bhh_r = (const float*)d_in[11];
  const float* h0r   = (const float*)d_in[12];
  const float* c0r   = (const float*)d_in[13];
  const float* Wfc   = (const float*)d_in[14];
  const float* bfc   = (const float*)d_in[15];
  float* out = (float*)d_out;

  // bump allocator over workspace
  size_t off = 0;
  auto alloc = [&](size_t bytes) -> void* {
    off = (off + 255) & ~(size_t)255;
    void* pp = (char*)d_ws + off;
    off += bytes;
    return pp;
  };

  unsigned short* xe_bf   = (unsigned short*)alloc((size_t)BB * SS * 224 * 2);   // [6400][224]
  unsigned short* Wihf_bf = (unsigned short*)alloc((size_t)NPAD * 224 * 2);
  unsigned short* Whhf_bf = (unsigned short*)alloc((size_t)NPAD * 256 * 2);
  unsigned short* Wihr_bf = (unsigned short*)alloc((size_t)NPAD * 1824 * 2);
  unsigned short* Whhr_bf = (unsigned short*)alloc((size_t)NPAD * 256 * 2);
  unsigned short* wpk     = (unsigned short*)alloc((size_t)6 * 16 * 448 * 2);
  unsigned short* ycnn_bf = (unsigned short*)alloc((size_t)NCNNP * 1824 * 2);
  float* bsum_f = (float*)alloc(G4H * 4);
  float* bsum_r = (float*)alloc(G4H * 4);
  float* zpre_f = (float*)alloc((size_t)BB * SS * G4H * 4);      // [6400][900]
  float* zpre_r = (float*)alloc((size_t)NCNN * G4H * 4);         // [6176][900]
  float* zh     = (float*)alloc((size_t)BB * G4H * 4);           // [32][900]
  float* c_st   = (float*)alloc((size_t)BB * HH * 4);
  unsigned short* hbf = (unsigned short*)alloc((size_t)64 * 256 * 2);  // padded rows
  float* hf = (float*)alloc((size_t)BB * SS * HH * 4);           // [32][200][225]
  float* hr = (float*)alloc((size_t)BB * CLIPS * HH * 4);        // [32][193][225]
  float* stats = (float*)alloc((size_t)6 * 16 * 2 * 4);
  float* norm  = (float*)alloc((size_t)6 * 16 * 2 * 4);
  float* convA = (float*)alloc((size_t)NPOS * 16 * 4);           // holds z1(4), z3(16), z5(4)
  float* convB = (float*)alloc((size_t)NPOS * 8 * 4);            // holds z2(8), z4(8), z6(1)

  // --- preprocessing ---
  zero_f<<<1, 256, 0, stream>>>(stats, 6 * 16 * 2);
  {
    long long n = (long long)BB * SS * 224;
    embed_bf<<<(unsigned)((n + 255) / 256), 256, 0, stream>>>(x, emb, xe_bf);
  }
  {
    long long t;
    t = (long long)NPAD * 224;  cvt_pad<<<(unsigned)((t + 255) / 256), 256, 0, stream>>>(Wih_f, Wihf_bf, G4H, 200, 224, t);
    t = (long long)NPAD * 256;  cvt_pad<<<(unsigned)((t + 255) / 256), 256, 0, stream>>>(Whh_f, Whhf_bf, G4H, 225, 256, t);
    t = (long long)NPAD * 1824; cvt_pad<<<(unsigned)((t + 255) / 256), 256, 0, stream>>>(Wih_r, Wihr_bf, G4H, 1800, 1824, t);
    t = (long long)NPAD * 256;  cvt_pad<<<(unsigned)((t + 255) / 256), 256, 0, stream>>>(Whh_r, Whhr_bf, G4H, 225, 256, t);
  }
  static const int CoL[6] = {4, 8, 16, 8, 4, 1};
  static const int CiL[6] = {1, 4, 8, 16, 8, 4};
  static const int KpL[6] = {32, 128, 224, 448, 224, 128};
  for (int l = 0; l < 6; ++l) {
    const float* w = (const float*)d_in[16 + 3 * l];
    pack_w<<<(16 * 448 + 255) / 256, 256, 0, stream>>>(w, wpk + (size_t)l * 16 * 448,
                                                       CoL[l], CiL[l] * 27);
  }
  bias_sum<<<(G4H + 255) / 256, 256, 0, stream>>>(bih_f, bhh_f, bsum_f, G4H);
  bias_sum<<<(G4H + 255) / 256, 256, 0, stream>>>(bih_r, bhh_r, bsum_r, G4H);

  // --- forward LSTM: hoisted input projection then recurrence ---
  gemm_bf16_nt<<<dim3(BB * SS / 64, NPAD / 64), 128, 0, stream>>>(xe_bf, Wihf_bf, bsum_f, zpre_f,
                                                                  BB * SS, G4H, 224, G4H);
  lstm_init<<<(64 * 256 + 255) / 256, 256, 0, stream>>>(h0f, c0f, c_st, hbf);
  for (int t = 0; t < SS; ++t) {
    gemm_bf16_nt<<<dim3(1, NPAD / 64), 128, 0, stream>>>(hbf, Whhf_bf, (const float*)nullptr, zh,
                                                         BB, G4H, 256, G4H);
    lstm_step<<<(BB * HH + 255) / 256, 256, 0, stream>>>(zpre_f, zh, c_st, hbf, hf, t, SS);
  }

  // --- 6 conv layers (implicit-GEMM WMMA) with BN fused into next layer's gather ---
  float* zbuf[6] = {convA, convB, convA, convB, convA, convB};
  unsigned conv_grid = (unsigned)(NPOS / 64);
  for (int l = 0; l < 6; ++l) {
    const float* zin = (l == 0) ? (const float*)nullptr : zbuf[l - 1];
    const float* nrm = (l == 0) ? (const float*)nullptr : (norm + (l - 1) * 32);
    conv3d_wmma<<<conv_grid, 128, 0, stream>>>(zin, hf, nrm, wpk + (size_t)l * 16 * 448,
                                               zbuf[l], CiL[l], CoL[l], KpL[l], l == 0 ? 1 : 0);
    bn_stats<<<dim3(512, CoL[l]), 256, 0, stream>>>(zbuf[l], stats + l * 32);
    bn_finalize<<<1, 16, 0, stream>>>(stats + l * 32, (const float*)d_in[17 + 3 * l],
                                      (const float*)d_in[18 + 3 * l], norm + l * 32, CoL[l]);
  }

  // --- BN + tanh + reshape for reverse LSTM input, hoisted projection, recurrence ---
  {
    long long n = (long long)NCNNP * 1824;
    pack_ycnn<<<(unsigned)((n + 255) / 256), 256, 0, stream>>>(zbuf[5], norm + 5 * 32, ycnn_bf);
  }
  gemm_bf16_nt<<<dim3(NCNNP / 64, NPAD / 64), 128, 0, stream>>>(ycnn_bf, Wihr_bf, bsum_r, zpre_r,
                                                                NCNN, G4H, 1824, G4H);
  lstm_init<<<(64 * 256 + 255) / 256, 256, 0, stream>>>(h0r, c0r, c_st, hbf);
  for (int t = 0; t < CLIPS; ++t) {
    gemm_bf16_nt<<<dim3(1, NPAD / 64), 128, 0, stream>>>(hbf, Whhr_bf, (const float*)nullptr, zh,
                                                         BB, G4H, 256, G4H);
    lstm_step<<<(BB * HH + 255) / 256, 256, 0, stream>>>(zpre_r, zh, c_st, hbf, hr, t, CLIPS);
  }

  // --- final FC ---
  fc_out<<<(NCNN + 255) / 256, 256, 0, stream>>>(hr, Wfc, bfc, out);
}
